// CHST_17635135717380
// MI455X (gfx1250) — compile-verified, matrix operand order checked by reading, not run on hardware
//
#include <hip/hip_runtime.h>

typedef __attribute__((ext_vector_type(16))) _Float16 v16h;
typedef __attribute__((ext_vector_type(8)))  _Float16 v8h;
typedef __attribute__((ext_vector_type(4)))  _Float16 v4h;
typedef __attribute__((ext_vector_type(8)))  float    v8f;
typedef __attribute__((ext_vector_type(4)))  float    v4f;

#define DEV static __device__ __forceinline__

#define EPSF  1e-15f
#define MAXNF (1.0f - 1e-5f)
#define HD    128
#define KNB   16
#define TILE  16

DEV float artanh_(float x) {
    x = fminf(fmaxf(x, -1.f + 1e-7f), 1.f - 1e-7f);
    return 0.5f * logf((1.f + x) / (1.f - x));
}
DEV float sigmoid_(float x) { return 1.f / (1.f + expf(-x)); }

DEV float wave_sum(float v) {
    v += __shfl_xor(v, 1);  v += __shfl_xor(v, 2);  v += __shfl_xor(v, 4);
    v += __shfl_xor(v, 8);  v += __shfl_xor(v, 16);
    return v;
}
DEV float half_sum(float v) {   // reduce within 16-lane half (wave32)
    v += __shfl_xor(v, 1);  v += __shfl_xor(v, 2);
    v += __shfl_xor(v, 4);  v += __shfl_xor(v, 8);
    return v;
}
DEV float half_max(float v) {
    v = fmaxf(v, __shfl_xor(v, 1)); v = fmaxf(v, __shfl_xor(v, 2));
    v = fmaxf(v, __shfl_xor(v, 4)); v = fmaxf(v, __shfl_xor(v, 8));
    return v;
}

// ---- Row layout: lane l holds dims {4l..4l+3} of each 128-dim block -------
template<int NB> struct Row { v4f b[NB]; };

template<int NB> DEV float row_dot(const Row<NB>& a, const Row<NB>& c) {
    float s = 0.f;
#pragma unroll
    for (int i = 0; i < NB; i++) { v4f p = a.b[i] * c.b[i]; s += p[0] + p[1] + p[2] + p[3]; }
    return wave_sum(s);
}
template<int NB> DEV float blk_norm(const Row<NB>& a, int i) {
    v4f p = a.b[i] * a.b[i];
    float n2 = wave_sum(p[0] + p[1] + p[2] + p[3]);
    return fmaxf(sqrtf(fmaxf(n2, 0.f)), EPSF);
}
// project with analytically-known norm^2 (updates rn2); no extra reduction
template<int NB> DEV void row_proj_n(Row<NB>& r, float& rn2) {
    float n = sqrtf(fmaxf(rn2, 0.f));
    if (n > MAXNF) {
        float s = MAXNF / n;
#pragma unroll
        for (int i = 0; i < NB; i++) r.b[i] *= s;
        rn2 = MAXNF * MAXNF;
    }
}
// mobius_add with known |x|^2, |y|^2; result norm^2 returned analytically.
// Costs exactly one wave reduction (x.y).
template<int NB> DEV Row<NB> row_madd_n(const Row<NB>& x, const Row<NB>& y,
                                        float x2, float y2, float& rn2) {
    float xy = row_dot(x, y);
    float A = 1.f + 2.f * xy + y2;
    float B = 1.f - x2;
    float inv = 1.f / fmaxf(1.f + 2.f * xy + x2 * y2, EPSF);
    Row<NB> r;
#pragma unroll
    for (int i = 0; i < NB; i++) r.b[i] = (A * x.b[i] + B * y.b[i]) * inv;
    rn2 = fmaxf((A * A * x2 + B * B * y2 + 2.f * A * B * xy) * inv * inv, 0.f);
    row_proj_n(r, rn2);
    return r;
}
// mobius_matvec tail: needs one reduction (|mx|); result norm = min(tanh,maxn)
template<int NB> DEV Row<NB> row_mmv_n(const Row<NB>& mx, float xn, float& rn2) {
    float mxn = fmaxf(sqrtf(fmaxf(row_dot(mx, mx), 0.f)), EPSF);
    float t = fminf(tanhf(mxn / xn * artanh_(xn)), MAXNF);
    float sc = t / mxn;
    Row<NB> r;
#pragma unroll
    for (int i = 0; i < NB; i++) r.b[i] = mx.b[i] * sc;
    rn2 = t * t;
    return r;
}
// mobius_pointwise_mul with known |x|; one reduction (|w*x|)
template<int NB> DEV Row<NB> row_mpm_xn(const Row<NB>& w, const Row<NB>& x,
                                        float xn, float& rn2) {
    Row<NB> wx;
#pragma unroll
    for (int i = 0; i < NB; i++) wx.b[i] = w.b[i] * x.b[i];
    float wxn = fmaxf(sqrtf(fmaxf(row_dot(wx, wx), 0.f)), EPSF);
    float t = fminf(tanhf(wxn / xn * artanh_(xn)), MAXNF);
    float sc = t / wxn;
#pragma unroll
    for (int i = 0; i < NB; i++) wx.b[i] *= sc;
    rn2 = t * t;
    return wx;
}
template<int NB> DEV Row<NB> row_mpm2(const Row<NB>& w, const Row<NB>& x, float& rn2) {
    float xn = fmaxf(sqrtf(fmaxf(row_dot(x, x), 0.f)), EPSF);
    return row_mpm_xn(w, x, xn, rn2);
}
template<int NB> DEV Row<NB> row_load(const float* base, int lane) {
    Row<NB> r;
#pragma unroll
    for (int i = 0; i < NB; i++) r.b[i] = *(const v4f*)(base + i * HD + 4 * lane);
    return r;
}
template<int NB> DEV void row_store(float* base, const Row<NB>& r, int lane) {
#pragma unroll
    for (int i = 0; i < NB; i++) *(v4f*)(base + i * HD + 4 * lane) = r.b[i];
}
DEV Row<1> row_load_h(const _Float16* base, int lane) {
    v4h t = *(const v4h*)(base + 4 * lane);
    Row<1> r;
    r.b[0] = (v4f){(float)t[0], (float)t[1], (float)t[2], (float)t[3]};
    return r;
}
DEV void row_store_h(_Float16* base, const Row<1>& r, int lane) {
    v4h t = {(_Float16)r.b[0][0], (_Float16)r.b[0][1], (_Float16)r.b[0][2], (_Float16)r.b[0][3]};
    *(v4h*)(base + 4 * lane) = t;
}

// ---- WMMA helpers ----------------------------------------------------------
DEV v8f wmma_(v16h a, v16h b, v8f c) {
    return __builtin_amdgcn_wmma_f32_16x16x32_f16(false, a, false, b, (short)0, c, false, false);
}
DEV v16h cvt_afrag(const float* rowp, int kb, int hi, float& ss) {
    const float* p0 = rowp + kb + hi * 8;
    v16h a;
#pragma unroll
    for (int j = 0; j < 8; j++) {
        float f0 = p0[j], f1 = p0[16 + j];
        ss += f0 * f0 + f1 * f1;
        a[j] = (_Float16)f0; a[j + 8] = (_Float16)f1;
    }
    return a;
}
DEV v16h load_bfrag(const _Float16* wrow, int kb, int hi) {
    const _Float16* p = wrow + kb + hi * 8;
    v8h lo = *(const v8h*)(p);
    v8h hh = *(const v8h*)(p + 16);
    v16h b;
#pragma unroll
    for (int j = 0; j < 8; j++) { b[j] = lo[j]; b[j + 8] = hh[j]; }
    return b;
}
// C[16, 16*tc] = A[16,128] (f32 rows) x W^T (f16 [Hout][128]), tiles [t0,t0+tc)
// Returns |A row (lane&15)|^2 (full 128 dims).
template<typename T>
DEV float wave_gemm(const float* Abase, int astride, const _Float16* W,
                    int t0, int tc, T* dst, int dstride, int lane) {
    int mr = lane & 15, hi = lane >> 4;
    const float* rowp = Abase + mr * astride;
    float ss = 0.f;
    v16h a0 = cvt_afrag(rowp, 0,  hi, ss);
    v16h a1 = cvt_afrag(rowp, 32, hi, ss);
    v16h a2 = cvt_afrag(rowp, 64, hi, ss);
    v16h a3 = cvt_afrag(rowp, 96, hi, ss);
    float xn2 = ss + __shfl_xor(ss, 16);
    int rofs = hi * 8;
    for (int t = t0; t < t0 + tc; ++t) {
        const _Float16* wrow = W + (t * 16 + mr) * HD;
        v8f c = {};
        c = wmma_(a0, load_bfrag(wrow, 0,  hi), c);
        c = wmma_(a1, load_bfrag(wrow, 32, hi), c);
        c = wmma_(a2, load_bfrag(wrow, 64, hi), c);
        c = wmma_(a3, load_bfrag(wrow, 96, hi), c);
        T* dp = dst + t * 16 + mr;
#pragma unroll
        for (int g = 0; g < 8; ++g) dp[(g + rofs) * dstride] = (T)c[g];
    }
    return xn2;
}

// ---- weight f32 -> f16 conversion -----------------------------------------
__global__ void cvt_weights(const float* __restrict__ Uiou, const float* __restrict__ Umso,
                            const float* __restrict__ Uf,   const float* __restrict__ Wq,
                            const float* __restrict__ Wk,   const float* __restrict__ Wc,
                            _Float16* __restrict__ out) {
    int i = blockIdx.x * blockDim.x + threadIdx.x;
    const int S = 16384;                          // 128*128
    if (i < S)                    out[i] = (_Float16)Wq[i];
    else if (i < 2 * S)           out[i] = (_Float16)Wk[i - S];
    else if (i < 3 * S)           out[i] = (_Float16)Wc[i - 2 * S];
    else if (i < 4 * S)           out[i] = (_Float16)Uf[i - 3 * S];
    else if (i < 4 * S + 32768)   out[i] = (_Float16)Uiou[i - 4 * S];
    else if (i < 4 * S + 81920)   out[i] = (_Float16)Umso[i - 4 * S - 32768];
}

// ---- fused hyperbolic tree-LSTM cell --------------------------------------
__global__ __launch_bounds__(128)
__attribute__((amdgpu_waves_per_eu(4)))
void hyp_tree_kernel(const float* __restrict__ x,       const float* __restrict__ f,
                     const float* __restrict__ iou1,    const float* __restrict__ mso1,
                     const float* __restrict__ mail_h1, const float* __restrict__ mail_c1,
                     const float* __restrict__ mail_x1, const float* __restrict__ del_t,
                     const float* __restrict__ b_q,     const float* __restrict__ b_k,
                     const float* __restrict__ b_c,     const float* __restrict__ b_iou,
                     const float* __restrict__ b_mso,
                     const float* __restrict__ a_param, const float* __restrict__ b_param,
                     const _Float16* __restrict__ Wq16, const _Float16* __restrict__ Wk16,
                     const _Float16* __restrict__ Wc16, const _Float16* __restrict__ Uf16,
                     const _Float16* __restrict__ Uiou16, const _Float16* __restrict__ Umso16,
                     float* __restrict__ out, int N) {
    __shared__ float smem[14464];                  // 57.86 KB
    const int lane = threadIdx.x & 31;
    const int wave = threadIdx.x >> 5;
    const int n0   = blockIdx.x * TILE;
    if (n0 + TILE > N) return;

    float* sC    = smem;                  // [4][16*128] per-wave GEMM staging
    float* sCw   = sC + wave * 2048;
    float* s_xq  = smem + 8192;           // x_q   [16][128]
    float* s_ht  = smem + 10240;          // exp0x then h_tild [16][128]
    float* s_cr  = smem + 12288;          // c_red [16][128]
    float* s_sc  = smem + 14336;          // 64 scalar slots
    float* s_ckn = smem + 14400;          // [4][16] |c_k_tilde|^2 per wave
    const int SC_XN = 0, SC_XQ2 = 16, SC_HT = 32, SC_CR = 48;

    // ---------- phase 1a: exp0x = expmap0(x) ----------
#pragma unroll 1
    for (int q = 0; q < 4; q++) {
        int r = wave * 4 + q; long n = n0 + r;
        Row<1> xr = row_load<1>(x + n * HD, lane);
        float un = fmaxf(sqrtf(fmaxf(row_dot(xr, xr), 0.f)), EPSF);
        float t = fminf(tanhf(un), MAXNF);
        Row<1> e; e.b[0] = xr.b[0] * (t / un);
        row_store<1>(s_ht + r * HD, e, lane);
        if (lane == 0) s_sc[SC_XN + r] = fmaxf(t, EPSF);
    }
    __syncthreads();

    // ---------- phase 1b: mq = exp0x @ Wq^T (waves split the 8 N-tiles) ----------
    wave_gemm<float>(s_ht, HD, Wq16, wave * 2, 2, s_xq, HD, lane);
    __syncthreads();

    // ---------- phase 1c: x_q = mobius_add(mobius_matvec(Wq, exp0x), b_q) ----------
    {
        Row<1> bq = row_load<1>(b_q, lane);
        float bq2 = row_dot(bq, bq);
#pragma unroll 1
        for (int q = 0; q < 4; q++) {
            int r = wave * 4 + q;
            Row<1> mx = row_load<1>(s_xq + r * HD, lane);
            float r2;
            Row<1> xq = row_mmv_n(mx, s_sc[SC_XN + r], r2);
            float q2;
            xq = row_madd_n(xq, bq, r2, bq2, q2);
            row_store<1>(s_xq + r * HD, xq, lane);
            if (lane == 0) s_sc[SC_XQ2 + r] = q2;
        }
    }
    __syncthreads();

    // ---------- phase 2: per-node mailbox reduction (each wave: 4 nodes) ----------
#pragma unroll 1
    for (int q = 0; q < 4; q++) {
        const int  local = wave * 4 + q;
        const long n     = n0 + local;
        const float* mh_base = mail_h1 + n * (long)(KNB * HD);
        const float* mc_base = mail_c1 + n * (long)(KNB * HD);

        if (q < 3) {   // prefetch next node's mailbox tiles (8KB each)
            __builtin_prefetch(mh_base + KNB * HD + lane * 64, 0, 3);
            __builtin_prefetch(mc_base + KNB * HD + lane * 64, 0, 3);
        }

        // --- (a) mhk = mail_h1 @ Wk^T ; HyperAttn scores ---
        float xh2 = wave_gemm<float>(mh_base, HD, Wk16, 0, 8, sCw, HD, lane);
        float xnh = fmaxf(sqrtf(fmaxf(xh2, 0.f)), EPSF);            // |mail_h1 row (lane&15)|
        // bias/query rows loaded AFTER the GEMM: short live ranges, L0-resident
        Row<1> bk = row_load<1>(b_k, lane);
        float bk2 = row_dot(bk, bk);
        Row<1> xqr = row_load<1>(s_xq + local * HD, lane);
        float  xq2 = s_sc[SC_XQ2 + local];
        float  dk  = 0.f;
#pragma unroll 1
        for (int k = 0; k < KNB; k++) {
            Row<1> mrow = row_load<1>(sCw + k * HD, lane);
            float xnk = __shfl(xnh, k);
            float h2;
            Row<1> hk = row_mmv_n(mrow, xnk, h2);
            hk = row_madd_n(hk, bk, h2, bk2, h2);
            float xy  = -row_dot(hk, xqr);
            float A   = 1.f + 2.f * xy + xq2;
            float B   = 1.f - h2;
            float den = fmaxf(1.f + 2.f * xy + h2 * xq2, EPSF);
            float n2  = (A * A * h2 + B * B * xq2 + 2.f * A * B * xy) / (den * den);
            float nv  = fminf(fmaxf(sqrtf(fmaxf(n2, 0.f)), EPSF), MAXNF);
            float d   = 2.f * artanh_(nv);
            if ((lane & 15) == k) dk = d;
        }
        // g_t computed here (not live across the GEMM)
        float gv = b_param[0] * expf(-a_param[0] * del_t[n * KNB + (lane & 15)]);
        // softmax(-d) * g_t  (per 16-lane half)
        float sneg = -dk;
        float smax = half_max(sneg);
        float e    = expf(sneg - smax);
        float score = (e / half_sum(e)) * gv;
        // gyromidpoint weights (scalar form of mobius_pointwise_mul)
        float tt  = fminf(tanhf(score * artanh_(xnh)), MAXNF);      // |h_scaled|
        float u   = tt / xnh;
        float lam = 2.f / fmaxf(1.f - tt * tt, EPSF);
        float wk  = lam * u;
        float dsum = half_sum(lam - 1.f);
        dsum = (fabsf(dsum) < EPSF) ? EPSF : dsum;
        // numer = sum_k w_k * mail_h1[n,k]
        v4f acc = {0.f, 0.f, 0.f, 0.f};
#pragma unroll 4
        for (int k = 0; k < KNB; k++) {
            float wkk = __shfl(wk, k);
            v4f mr = *(const v4f*)(mh_base + k * HD + 4 * lane);
            acc += wkk * mr;
        }
        Row<1> ht; ht.b[0] = acc * (1.f / dsum);
        float an = fmaxf(sqrtf(fmaxf(row_dot(ht, ht), 0.f)), EPSF);
        float t5 = fminf(tanhf(0.5f * artanh_(an)), MAXNF);         // mobius_scalar_mul(0.5)
        ht.b[0] *= (t5 / an);
        row_store<1>(s_ht + local * HD, ht, lane);
        if (lane == 0) s_sc[SC_HT + local] = fmaxf(t5, EPSF);

        // --- (b) c-chain: staged as f16 tiles in the wave's 8KB LDS region ---
        _Float16* sH0 = (_Float16*)sCw;           // [16][128] f16: mc then mf
        _Float16* sH1 = (_Float16*)(sCw + 1024);  // [16][128] f16: c_k_tilde
        float xc2 = wave_gemm<_Float16>(mc_base, HD, Wc16, 0, 8, sH0, HD, lane);
        float xnc = fmaxf(sqrtf(fmaxf(xc2, 0.f)), EPSF);
        Row<1> bc = row_load<1>(b_c, lane);       // after GEMM: short live range
        float bc2 = row_dot(bc, bc);
#pragma unroll 1
        for (int k = 0; k < KNB; k++) {
            Row<1> mc = row_load_h(sH0 + k * HD, lane);
            float xck = __shfl(xnc, k);
            float w2;
            Row<1> wv = row_mmv_n(mc, xck, w2);
            wv = row_madd_n(wv, bc, w2, bc2, w2);
            // c_sk = expmap0(tanh(logmap0(wv)))
            float wn = fmaxf(sqrtf(w2), EPSF);
            float lml = artanh_(wn) / wn;
            Row<1> z;
#pragma unroll
            for (int j = 0; j < 4; j++) z.b[0][j] = tanhf(lml * wv.b[0][j]);
            float zn = fmaxf(sqrtf(fmaxf(row_dot(z, z), 0.f)), EPSF);
            float te = fminf(tanhf(zn), MAXNF);
            Row<1> csk; csk.b[0] = z.b[0] * (te / zn);
            float csk2 = te * te;
            float cskn = fmaxf(te, EPSF);
            // c_sk_hat = mpm(c_sk, g_t)  (scalar weight)
            float gk = __shfl(gv, k);
            float th = fminf(tanhf(gk * artanh_(cskn)), MAXNF);
            Row<1> chat; chat.b[0] = csk.b[0] * (th / cskn);
            float chat2 = th * th;
            // c_k_tilde = madd(madd(-c_sk, mail_c1), c_sk_hat)
            Row<1> c1r = row_load<1>(mc_base + k * HD, lane);
            float c12 = xck * xck;
            Row<1> neg; neg.b[0] = -csk.b[0];
            float t2_;
            Row<1> cTk = row_madd_n(neg, c1r, csk2, c12, t2_);
            float ck2;
            Row<1> ckt = row_madd_n(cTk, chat, t2_, chat2, ck2);
            row_store_h(sH1 + k * HD, ckt, lane);
            if (lane == 0) s_ckn[wave * 16 + k] = ck2;
        }
        // mf = mail_h1 @ Uf^T ; fgate ; weighted midpoint -> c_red
        wave_gemm<_Float16>(mh_base, HD, Uf16, 0, 8, sH0, HD, lane);
        Row<1> fr = row_load<1>(f + n * HD, lane);
        float fr2 = row_dot(fr, fr);
        v4f vacc = {0.f, 0.f, 0.f, 0.f};
        float dacc = 0.f;
#pragma unroll 1
        for (int k = 0; k < KNB; k++) {
            Row<1> mf = row_load_h(sH0 + k * HD, lane);
            float xnk = __shfl(xnh, k);
            float tmp2;
            Row<1> tmp = row_mmv_n(mf, xnk, tmp2);
            float fa2;
            Row<1> fa = row_madd_n(fr, tmp, fr2, tmp2, fa2);
            float fn = fmaxf(sqrtf(fa2), EPSF);
            float lml = artanh_(fn) / fn;
            Row<1> fg;
#pragma unroll
            for (int j = 0; j < 4; j++) fg.b[0][j] = sigmoid_(lml * fa.b[0][j]);
            Row<1> ck = row_load_h(sH1 + k * HD, lane);
            float ckn = fmaxf(sqrtf(fmaxf(s_ckn[wave * 16 + k], 0.f)), EPSF);
            float p2;
            Row<1> pk = row_mpm_xn(fg, ck, ckn, p2);
            float lamk = 2.f / fmaxf(1.f - p2, EPSF);
            vacc += lamk * pk.b[0];
            dacc += lamk - 1.f;
        }
        dacc = (fabsf(dacc) < EPSF) ? EPSF : dacc;
        Row<1> cr; cr.b[0] = vacc * (1.f / dacc);
        float crn = fmaxf(sqrtf(fmaxf(row_dot(cr, cr), 0.f)), EPSF);
        float tc5 = fminf(tanhf(0.5f * artanh_(crn)), MAXNF);
        cr.b[0] *= (tc5 / crn);
        row_store<1>(s_cr + local * HD, cr, lane);
        if (lane == 0) s_sc[SC_CR + local] = tc5 * tc5;

        // --- (c) x_out = (mean_k mail_x1 + x) / 2 ---
        v4f xs = {0.f, 0.f, 0.f, 0.f};
#pragma unroll 4
        for (int k = 0; k < KNB; k++)
            xs += *(const v4f*)(mail_x1 + n * (long)(KNB * HD) + k * HD + 4 * lane);
        v4f xr2 = *(const v4f*)(x + n * HD + 4 * lane);
        xs = (xs * (1.f / 16.f) + xr2) * 0.5f;
        *(v4f*)(out + (long)2 * N * HD + n * HD + 4 * lane) = xs;
    }
    __syncthreads();

    // ---------- phase 3: iou/mso GEMMs on h_tild (staging region reused) ----------
    float* iouL = smem;            // [16][256]
    float* msoL = smem + 4096;     // [16][384]
    wave_gemm<float>(s_ht, HD, Uiou16, wave * 4, 4, iouL, 256, lane);
    wave_gemm<float>(s_ht, HD, Umso16, wave * 6, 6, msoL, 384, lane);
    __syncthreads();

    {
        Row<2> bi = row_load<2>(b_iou, lane);  float bi2 = row_dot(bi, bi);
        Row<3> bm = row_load<3>(b_mso, lane);  float bm2 = row_dot(bm, bm);
#pragma unroll 1
        for (int q = 0; q < 4; q++) {
            const int  local = wave * 4 + q;
            const long n     = n0 + local;
            float htn = s_sc[SC_HT + local];

            Row<2> mi = row_load<2>(iouL + local * 256, lane);
            float si2;
            Row<2> si = row_mmv_n(mi, htn, si2);
            Row<2> i1 = row_load<2>(iou1 + n * 256, lane);
            float i12 = row_dot(i1, i1);
            float io2;
            Row<2> io = row_madd_n(i1, si, i12, si2, io2);
            io = row_madd_n(io, bi, io2, bi2, io2);
            float nI = blk_norm(io, 0), nU = blk_norm(io, 1);
            float lI = artanh_(nI) / nI, lU = artanh_(nU) / nU;
            Row<1> ig, ug;
#pragma unroll
            for (int j = 0; j < 4; j++) {
                ig.b[0][j] = sigmoid_(lI * io.b[0][j]);
                ug.b[0][j] = tanhf(lU * io.b[1][j]);
            }

            Row<3> mm = row_load<3>(msoL + local * 384, lane);
            float sm2;
            Row<3> sm = row_mmv_n(mm, htn, sm2);
            Row<3> m1 = row_load<3>(mso1 + n * 384, lane);
            float m12 = row_dot(m1, m1);
            float mo2;
            Row<3> mo = row_madd_n(m1, sm, m12, sm2, mo2);
            mo = row_madd_n(mo, bm, mo2, bm2, mo2);
            float nM = blk_norm(mo, 0), nS = blk_norm(mo, 1), nO = blk_norm(mo, 2);
            float lM = artanh_(nM) / nM, lS = artanh_(nS) / nS, lO = artanh_(nO) / nO;
            Row<1> mg, sg, og;
#pragma unroll
            for (int j = 0; j < 4; j++) {
                mg.b[0][j] = sigmoid_(lM * mo.b[0][j]);
                sg.b[0][j] = sigmoid_(lS * mo.b[1][j]);
                og.b[0][j] = sigmoid_(lO * mo.b[2][j]);
            }

            float t1n2, t2n2;
            Row<1> t1 = row_mpm2(ig, ug, t1n2);
            Row<1> t2 = row_mpm2(mg, sg, t2n2);
            Row<1> crd = row_load<1>(s_cr + local * HD, lane);
            float crd2 = s_sc[SC_CR + local];
            float cc2;
            Row<1> cellA = row_madd_n(t1, crd, t1n2, crd2, cc2);
            Row<1> cell  = row_madd_n(cellA, t2, cc2, t2n2, cc2);
            float cn = fmaxf(sqrtf(fmaxf(cc2, 0.f)), EPSF);
            float lc = artanh_(cn) / cn;
            Row<1> tl;
#pragma unroll
            for (int j = 0; j < 4; j++) tl.b[0][j] = tanhf(lc * cell.b[0][j]);
            float hn2;
            Row<1> hh = row_mpm2(og, tl, hn2);

            row_store<1>(out + n * HD, hh, lane);
            row_store<1>(out + (long)N * HD + n * HD, cell, lane);
        }
    }
}

extern "C" void kernel_launch(void* const* d_in, const int* in_sizes, int n_in,
                              void* d_out, int out_size, void* d_ws, size_t ws_size,
                              hipStream_t stream) {
    (void)n_in; (void)out_size; (void)ws_size;
    const float* x       = (const float*)d_in[0];
    const float* f       = (const float*)d_in[1];
    const float* iou1    = (const float*)d_in[2];
    const float* mso1    = (const float*)d_in[3];
    const float* mail_h1 = (const float*)d_in[4];
    const float* mail_c1 = (const float*)d_in[5];
    const float* mail_x1 = (const float*)d_in[6];
    const float* del_t   = (const float*)d_in[7];
    const float* U_iou   = (const float*)d_in[8];
    const float* U_mso   = (const float*)d_in[9];
    const float* U_f     = (const float*)d_in[10];
    const float* W_q     = (const float*)d_in[11];
    const float* b_q     = (const float*)d_in[12];
    const float* W_k     = (const float*)d_in[13];
    const float* b_k     = (const float*)d_in[14];
    const float* W_c     = (const float*)d_in[15];
    const float* b_c     = (const float*)d_in[16];
    const float* b_iou   = (const float*)d_in[17];
    const float* b_mso   = (const float*)d_in[18];
    const float* a_param = (const float*)d_in[19];
    const float* b_param = (const float*)d_in[20];

    const int N = in_sizes[0] / HD;   // 20000

    _Float16* W16 = (_Float16*)d_ws;  // f16 weights: Wq|Wk|Wc|Uf|Uiou|Umso
    const int total = 4 * 16384 + 32768 + 49152;   // 147456 elements
    cvt_weights<<<(total + 255) / 256, 256, 0, stream>>>(U_iou, U_mso, U_f, W_q, W_k, W_c, W16);

    hyp_tree_kernel<<<N / TILE, 128, 0, stream>>>(
        x, f, iou1, mso1, mail_h1, mail_c1, mail_x1, del_t,
        b_q, b_k, b_c, b_iou, b_mso, a_param, b_param,
        W16, W16 + 16384, W16 + 32768, W16 + 49152, W16 + 65536, W16 + 98304,
        (float*)d_out, N);
}
